// GNN_layer_10453950398768
// MI455X (gfx1250) — matrix operand order repaired and checked by reading, not compiled, not run on hardware
//
#include <hip/hip_runtime.h>

#define NG 16
#define NN 1024
#define D1 64
#define D2 64

typedef __attribute__((ext_vector_type(16))) __bf16 v16bf;
typedef __attribute__((ext_vector_type(8)))  float  v8f;

__device__ __forceinline__ unsigned f2bf(float f) {
    unsigned u = __float_as_uint(f);
    unsigned r = u + 0x7FFFu + ((u >> 16) & 1u);   // round-to-nearest-even
    return r >> 16;
}

// ---------------- Kernel 1: rowmean + diag (one wave32 per row) ----------------
__global__ void k1_rowstats(const float* __restrict__ A,
                            float* __restrict__ rowmean,
                            float* __restrict__ diag) {
    const int lane = threadIdx.x & 31;
    const int wid  = threadIdx.x >> 5;
    const int row  = blockIdx.x * 8 + wid;           // 0 .. NG*NN-1
    const float* ar = A + (size_t)row * NN;
    float s = 0.f;
#pragma unroll
    for (int k = 0; k < NN / 32; ++k) s += ar[lane + 32 * k];
#pragma unroll
    for (int off = 16; off > 0; off >>= 1) s += __shfl_xor(s, off, 32);
    if (lane == 0) {
        rowmean[row] = s * (1.0f / NN);
        diag[row]    = ar[row & (NN - 1)];
    }
}

// ---------------- Kernel 2: per-graph stats, pg vectors, per-node term ----------------
__global__ void k2_graphstats(const float* __restrict__ X,  const float* __restrict__ coeffs,
                              const float* __restrict__ wA1, const float* __restrict__ wA2,
                              const float* __restrict__ w12, const float* __restrict__ w15,
                              const float* __restrict__ w16,
                              const float* __restrict__ w22, const float* __restrict__ w25,
                              const float* __restrict__ w26,
                              const float* __restrict__ rowmean, const float* __restrict__ diag,
                              float* __restrict__ term, float* __restrict__ meanX,
                              float* __restrict__ basev, float* __restrict__ pg1,
                              float* __restrict__ pg2) {
    __shared__ float red[256];
    __shared__ float smx[D1];
    __shared__ float sscal[3];
    const int g = blockIdx.x, t = threadIdx.x;
    const float* rm = rowmean + g * NN;
    const float* dg = diag    + g * NN;
    const float* Xg = X + (size_t)g * NN * D1;

    // mean_diag
    float s = 0.f;
    for (int k = t; k < NN; k += 256) s += dg[k];
    red[t] = s; __syncthreads();
    for (int o = 128; o > 0; o >>= 1) { if (t < o) red[t] += red[t + o]; __syncthreads(); }
    if (t == 0) sscal[0] = red[0] * (1.0f / NN);
    __syncthreads();
    // mean_all
    s = 0.f;
    for (int k = t; k < NN; k += 256) s += rm[k];
    red[t] = s; __syncthreads();
    for (int o = 128; o > 0; o >>= 1) { if (t < o) red[t] += red[t + o]; __syncthreads(); }
    if (t == 0) sscal[1] = red[0] * (1.0f / NN);
    __syncthreads();
    // mean_X
    {
        const int e = t & 63, q = t >> 6;
        float sx = 0.f;
        for (int node = q; node < NN; node += 4) sx += Xg[node * D1 + e];
        red[t] = sx; __syncthreads();
        if (t < 64) {
            float m = (red[t] + red[t + 64] + red[t + 128] + red[t + 192]) * (1.0f / NN);
            smx[t] = m;
            meanX[g * D1 + t] = m;
        }
    }
    __syncthreads();
    // p2 + base
    if (t == 0) {
        float p2 = 0.f;
        for (int d = 0; d < D1; ++d) p2 += smx[d] * wA2[d];
        float b = coeffs[1] * sscal[1] + coeffs[2] * sscal[0] + p2;
        sscal[2] = b;
        basev[g] = b;
    }
    __syncthreads();
    const float mdiag = sscal[0], mall = sscal[1];
    // pg1, pg2
    if (t < D2) {
        float a1 = mdiag * w15[t] + mall * w16[t];
        float a2 = mdiag * w25[t] + mall * w26[t];
        for (int d = 0; d < D1; ++d) {
            a1 += smx[d] * w12[t * D1 + d];
            a2 += smx[d] * w22[t * D1 + d];
        }
        pg1[g * D2 + t] = a1;
        pg2[g * D2 + t] = a2;
    }
    // term[x] = c3*rowmean + c4*diag + p1
    const float c3 = coeffs[3], c4 = coeffs[4];
    for (int node = t; node < NN; node += 256) {
        const float* xr = Xg + node * D1;
        float p = 0.f;
#pragma unroll 8
        for (int d = 0; d < D1; ++d) p += xr[d] * wA1[d];
        term[g * NN + node] = c3 * rm[node] + c4 * dg[node] + p;
    }
}

// ---------------- Kernel 3: X1_t (fp32) and X2_t transposed bf16 ----------------
__global__ void k3_xtrans(const float* __restrict__ X,
                          const float* __restrict__ w11, const float* __restrict__ w13,
                          const float* __restrict__ w14,
                          const float* __restrict__ w21, const float* __restrict__ w23,
                          const float* __restrict__ w24,
                          const float* __restrict__ rowmean, const float* __restrict__ diag,
                          const float* __restrict__ pg1, const float* __restrict__ pg2,
                          float* __restrict__ X1, unsigned short* __restrict__ X2T) {
    __shared__ float w1s[D2 * D1];
    __shared__ float w2s[D2 * D1];
    __shared__ float Xs[64 * D1];
    const int g  = blockIdx.x >> 4;
    const int nb = (blockIdx.x & 15) * 64;
    const int t  = threadIdx.x;
    for (int i = t; i < D2 * D1; i += 256) { w1s[i] = w11[i]; w2s[i] = w21[i]; }
    for (int i = t; i < 64 * D1; i += 256) Xs[i] = X[((size_t)g * NN + nb) * D1 + i];
    __syncthreads();
    const int node  = t >> 2;
    const int e0    = (t & 3) * 16;
    const int gnode = nb + node;
    const float rm = rowmean[g * NN + gnode];
    const float dg = diag   [g * NN + gnode];
    const float* xr = Xs + node * D1;
    for (int ee = 0; ee < 16; ++ee) {
        const int e = e0 + ee;
        float a1 = rm * w13[e] + dg * w14[e] + pg1[g * D2 + e];
        float a2 = rm * w23[e] + dg * w24[e] + pg2[g * D2 + e];
        const float* r1 = w1s + e * D1;
        const float* r2 = w2s + e * D1;
#pragma unroll 8
        for (int d = 0; d < D1; ++d) { a1 += xr[d] * r1[d]; a2 += xr[d] * r2[d]; }
        X1[((size_t)g * NN + gnode) * D2 + e] = a1;
        X2T[((size_t)g * D2 + e) * NN + gnode] = (unsigned short)f2bf(a2);
    }
}

// ---------------- Kernel 4: fused A_t generation + WMMA GEMM (K-chunk = 64) ----------------
#define KP 72   // LDS row pitch in ushorts (64 + 8 pad); 144B, keeps 16B alignment

__global__ void __launch_bounds__(256)
k4_fused(const float* __restrict__ A, const float* __restrict__ coeffs,
         const float* __restrict__ term, const float* __restrict__ basev,
         const float* __restrict__ X1, const unsigned short* __restrict__ X2T,
         float* __restrict__ At, float* __restrict__ Out) {
    __shared__ __align__(16) unsigned short sA[64 * KP];   // 64 rows x 64 k (bf16 A_t)
    __shared__ __align__(16) unsigned short sB[64 * KP];   // 64 e   x 64 k (bf16 X2T)

    const int g       = blockIdx.x >> 4;
    const int rowbase = (blockIdx.x & 15) * 64;
    const int t       = threadIdx.x;
    const float c0    = coeffs[0];
    const float* termg = term + g * NN;

    const int arow = t >> 2;              // 0..63  (row for A, e for B)
    const int acol = (t & 3) * 16;        // 0,16,32,48
    const float* Abase  = A  + (((size_t)g * NN + rowbase + arow) * NN) + acol;
    float*       Atbase = At + (((size_t)g * NN + rowbase + arow) * NN) + acol;
    const unsigned short* Bsrc = X2T + ((size_t)g * D2 + arow) * NN + acol;

    // loop-invariant row term, read once from global (not LDS -> hoistable)
    const float rterm = basev[g] + termg[rowbase + arow];

    const int lane = t & 31;
    const int w    = t >> 5;
    const int rt   = w >> 1;              // row tile 0..3
    const int ctb  = (w & 1) * 2;         // col tile base 0 or 2
    const int m    = lane & 15;
    const int half = lane >> 4;
    const int rrow = (rt * 16 + m) * KP;
    const int be0  = (ctb * 16 + m) * KP;
    const int be1  = ((ctb + 1) * 16 + m) * KP;

    v8f acc0 = {}; v8f acc1 = {};

    for (int k0 = 0; k0 < NN; k0 += 64) {
        __syncthreads();                                   // LDS consumed by prev wmma
        float4 a[4], tv[4], o[4];
#pragma unroll
        for (int q = 0; q < 4; ++q) {
            a[q]  = *(const float4*)(Abase + k0 + 4 * q);
            tv[q] = *(const float4*)(termg + k0 + acol + 4 * q);
        }
        if (k0 + 64 < NN) __builtin_prefetch(Abase + k0 + 64, 0, 1);
#pragma unroll
        for (int q = 0; q < 4; ++q) {
            o[q].x = fmaf(c0, a[q].x, rterm + tv[q].x);
            o[q].y = fmaf(c0, a[q].y, rterm + tv[q].y);
            o[q].z = fmaf(c0, a[q].z, rterm + tv[q].z);
            o[q].w = fmaf(c0, a[q].w, rterm + tv[q].w);
            *(float4*)(Atbase + k0 + 4 * q) = o[q];        // A_t fp32 output
        }
        uint4 pk0, pk1;                                    // bf16 pack for WMMA
        pk0.x = f2bf(o[0].x) | (f2bf(o[0].y) << 16);
        pk0.y = f2bf(o[0].z) | (f2bf(o[0].w) << 16);
        pk0.z = f2bf(o[1].x) | (f2bf(o[1].y) << 16);
        pk0.w = f2bf(o[1].z) | (f2bf(o[1].w) << 16);
        pk1.x = f2bf(o[2].x) | (f2bf(o[2].y) << 16);
        pk1.y = f2bf(o[2].z) | (f2bf(o[2].w) << 16);
        pk1.z = f2bf(o[3].x) | (f2bf(o[3].y) << 16);
        pk1.w = f2bf(o[3].z) | (f2bf(o[3].w) << 16);
        *(uint4*)&sA[arow * KP + acol]     = pk0;
        *(uint4*)&sA[arow * KP + acol + 8] = pk1;

        *(uint4*)&sB[arow * KP + acol]     = *(const uint4*)(Bsrc + k0);
        *(uint4*)&sB[arow * KP + acol + 8] = *(const uint4*)(Bsrc + k0 + 8);
        __syncthreads();

#pragma unroll
        for (int ks = 0; ks < 64; ks += 32) {
            union { uint4 u[2]; v16bf v; } fa, fb0, fb1;
            fa.u[0]  = *(const uint4*)&sA[rrow + ks + half * 16];       // K = ks+half*16+0..7
            fa.u[1]  = *(const uint4*)&sA[rrow + ks + half * 16 + 8];   // K = ks+half*16+8..15
            fb0.u[0] = *(const uint4*)&sB[be0 + ks + half * 16];        // K = ks+half*16+0..7
            fb0.u[1] = *(const uint4*)&sB[be0 + ks + half * 16 + 8];    // K = ks+half*16+8..15
            fb1.u[0] = *(const uint4*)&sB[be1 + ks + half * 16];
            fb1.u[1] = *(const uint4*)&sB[be1 + ks + half * 16 + 8];

            acc0 = __builtin_amdgcn_wmma_f32_16x16x32_bf16(false, fa.v, false, fb0.v,
                                                           (short)0, acc0, false, false);
            acc1 = __builtin_amdgcn_wmma_f32_16x16x32_bf16(false, fa.v, false, fb1.v,
                                                           (short)0, acc1, false, false);
        }
    }

    // Epilogue: out = acc/n + X1_t   (C layout: VGPR r -> M=r+8*half, N=lane%16)
    const float inv = 1.0f / NN;
    const int orow = rowbase + rt * 16 + half * 8;
    const int ecol = ctb * 16 + m;
#pragma unroll
    for (int r = 0; r < 8; ++r) {
        const size_t o0 = ((size_t)g * NN + orow + r) * D2 + ecol;
        Out[o0]      = fmaf(acc0[r], inv, X1[o0]);
        Out[o0 + 16] = fmaf(acc1[r], inv, X1[o0 + 16]);
    }
}

extern "C" void kernel_launch(void* const* d_in, const int* in_sizes, int n_in,
                              void* d_out, int out_size, void* d_ws, size_t ws_size,
                              hipStream_t stream) {
    const float* A      = (const float*)d_in[0];
    const float* X      = (const float*)d_in[1];
    const float* coeffs = (const float*)d_in[2];
    const float* wA1    = (const float*)d_in[3];
    const float* wA2    = (const float*)d_in[4];
    const float* w11    = (const float*)d_in[5];
    const float* w12    = (const float*)d_in[6];
    const float* w13    = (const float*)d_in[7];
    const float* w14    = (const float*)d_in[8];
    const float* w15    = (const float*)d_in[9];
    const float* w16    = (const float*)d_in[10];
    const float* w21    = (const float*)d_in[11];
    const float* w22    = (const float*)d_in[12];
    const float* w23    = (const float*)d_in[13];
    const float* w24    = (const float*)d_in[14];
    const float* w25    = (const float*)d_in[15];
    const float* w26    = (const float*)d_in[16];

    float* out  = (float*)d_out;
    float* At   = out;                               // (16,1024,1024)
    float* Outm = out + (size_t)NG * NN * NN;        // (16,1024,64)

    char* ws = (char*)d_ws;
    float* rowmean = (float*)ws;  ws += (size_t)NG * NN * 4;
    float* diag    = (float*)ws;  ws += (size_t)NG * NN * 4;
    float* term    = (float*)ws;  ws += (size_t)NG * NN * 4;
    float* meanX   = (float*)ws;  ws += (size_t)NG * D1 * 4;
    float* basev   = (float*)ws;  ws += 256;
    float* pg1     = (float*)ws;  ws += (size_t)NG * D2 * 4;
    float* pg2     = (float*)ws;  ws += (size_t)NG * D2 * 4;
    float* X1      = (float*)ws;  ws += (size_t)NG * NN * D2 * 4;
    unsigned short* X2T = (unsigned short*)ws;       // (16,64,1024) bf16

    hipLaunchKernelGGL(k1_rowstats, dim3(NG * NN / 8), dim3(256), 0, stream,
                       A, rowmean, diag);
    hipLaunchKernelGGL(k2_graphstats, dim3(NG), dim3(256), 0, stream,
                       X, coeffs, wA1, wA2, w12, w15, w16, w22, w25, w26,
                       rowmean, diag, term, meanX, basev, pg1, pg2);
    hipLaunchKernelGGL(k3_xtrans, dim3(NG * 16), dim3(256), 0, stream,
                       X, w11, w13, w14, w21, w23, w24,
                       rowmean, diag, pg1, pg2, X1, X2T);
    hipLaunchKernelGGL(k4_fused, dim3(NG * 16), dim3(256), 0, stream,
                       A, coeffs, term, basev, X1, X2T, At, Outm);
}
